// ExplanationNet_74277164417515
// MI455X (gfx1250) — matrix operand order repaired and checked by reading, not compiled, not run on hardware
//
#include <hip/hip_runtime.h>
#include <math.h>

typedef __attribute__((ext_vector_type(16))) __bf16 v16bf;
typedef __attribute__((ext_vector_type(8)))  float  v8f;
typedef unsigned int u32;
typedef __attribute__((ext_vector_type(4))) u32 v4u;
typedef __attribute__((ext_vector_type(8))) u32 v8u;
typedef __attribute__((ext_vector_type(8))) int v8i;
typedef __attribute__((ext_vector_type(4))) int v4i;

#define BN_SCALE 0.99999500003749969f   /* 1/sqrt(1+1e-5) */

#define NNODE 32768
#define NEDGE 262144
#define GNUM  512
#define BSZ   4096

enum { EP_NONE=0, EP_BIAS, EP_BIAS_RELU, EP_BIAS_BN_RELU, EP_BIAS_SIGMOID, EP_BIAS_RELU_BN };

#if defined(__has_builtin)
#if __has_builtin(__builtin_amdgcn_tensor_load_to_lds)
#define USE_TDM 1
#else
#define USE_TDM 0
#endif
#else
#define USE_TDM 0
#endif

#if USE_TDM
/* Tensor Data Mover: DMA a 2D fp32 tile (tile_d0 x tile_d1) from global
 * (row stride = stride_elems) into LDS at lds_byte_addr. OOB rows/cols
 * (beyond tens_d0 / tens_d1) are zero-filled by the TDM per ISA 8.2. */
__device__ __forceinline__ void tdm_load_2d(u32 lds_byte_addr, const void* gaddr,
                                            u32 tile_d0, u32 tile_d1,
                                            u32 tens_d0, u32 tens_d1,
                                            u32 stride_elems)
{
  unsigned long long ga = (unsigned long long)gaddr;
  v4u g0;
  g0[0] = 1u;                                       /* count=1, user desc */
  g0[1] = lds_byte_addr;                            /* lds_addr [63:32]   */
  g0[2] = (u32)ga;                                  /* global_addr[31:0]  */
  g0[3] = (u32)((ga >> 32) & 0x1FFFFFFu) | (2u << 30); /* addr[56:32] | type=2 */
  v8i g1;
  g1[0] = 0x00020000;                               /* data_size=2 (4B)   */
  g1[1] = (int)((tens_d0 & 0xFFFFu) << 16);         /* tensor_dim0 lo     */
  g1[2] = (int)((tens_d0 >> 16) | ((tens_d1 & 0xFFFFu) << 16));
  g1[3] = (int)((tens_d1 >> 16) | ((tile_d0 & 0xFFFFu) << 16));
  g1[4] = (int)(tile_d1 & 0xFFFFu);                 /* tile_dim1, tile_dim2=0 */
  g1[5] = (int)stride_elems;                        /* tensor_dim0_stride */
  g1[6] = 0;
  g1[7] = 0;
  v4i z4 = {0, 0, 0, 0};
#if __clang_major__ >= 23
  v8i z8 = {0, 0, 0, 0, 0, 0, 0, 0};
  __builtin_amdgcn_tensor_load_to_lds(g0, g1, z4, z4, z8, 0);
#else
  __builtin_amdgcn_tensor_load_to_lds(g0, g1, z4, z4, 0);
#endif
}
#endif

/* ------------------------------------------------------------------ */
/* WMMA bf16 GEMM: C[M,N] = epilogue(A[M,K] @ W[K,N])                 */
/* block = 4 wave32; tile 64x64; wave w -> rows [16w,16w+16) x 64 cols*/
/* TDM stages raw fp32 tiles to LDS; threads convert into per-lane    */
/* WMMA fragment layout so each fragment is 2x ds_load_b128.          */
/* ------------------------------------------------------------------ */
template<int EP>
__global__ __launch_bounds__(128) void k_gemm(
    const float* __restrict__ A, int lda,
    const float* __restrict__ W,            /* K x N, ld = N */
    float* __restrict__ C, int ldc,
    int M, int N, int K,
    const float* __restrict__ bias,
    const float* __restrict__ bng,
    const float* __restrict__ bnb)
{
  __shared__ float sAraw[64][32];                   /* raw fp32 A tile  */
  __shared__ float sBraw[32][64];                   /* raw fp32 W tile  */
  __shared__ __align__(32) u32 sAf[4][32][8];       /* A frags [rt][lane][dw] */
  __shared__ __align__(32) u32 sBf[4][32][8];       /* B frags [cb][lane][dw] */

  const int tid  = threadIdx.x;
  const int lane = tid & 31;
  const int wave = tid >> 5;
  const int half = lane >> 4;
  const int l16  = lane & 15;
  const int m0 = blockIdx.y * 64;
  const int n0 = blockIdx.x * 64;

  v8f acc[4] = {};

  for (int k0 = 0; k0 < K; k0 += 32) {
#if USE_TDM
    if (tid == 0) {
      tdm_load_2d((u32)(unsigned long long)&sAraw[0][0],
                  A + (long)m0*lda + k0, 32, 64,
                  (u32)(K - k0), (u32)(M - m0), (u32)lda);
      tdm_load_2d((u32)(unsigned long long)&sBraw[0][0],
                  W + (long)k0*N + n0, 64, 32,
                  (u32)(N - n0), (u32)(K - k0), (u32)N);
      __builtin_amdgcn_s_wait_tensorcnt(0);
    }
#else
    for (int i = tid; i < 64*32; i += 128) {
      int r = i >> 5, c = i & 31;
      int gk = k0 + c;
      sAraw[r][c] = (gk < K) ? A[(long)(m0 + r)*lda + gk] : 0.0f;
    }
    for (int i = tid; i < 32*64; i += 128) {
      int r = i >> 6, c = i & 63;
      int gk = k0 + r;
      sBraw[r][c] = (gk < K) ? W[(long)gk*N + (n0 + c)] : 0.0f;
    }
#endif
    __syncthreads();

    /* convert fp32 -> packed bf16 pairs scattered into fragment layout */
    #pragma unroll
    for (int p = tid; p < 1024; p += 128) {         /* A: 64 rows x 16 pairs */
      int r  = p >> 4;
      int Ke = (p & 15) << 1;
      int hf = (Ke >> 3) & 1;
      int kk = Ke - (hf << 3);
      int v  = ((kk & 15) >> 1) | ((kk >> 4) << 2);
      union { __bf16 h[2]; u32 u; } pk;
      pk.h[0] = (__bf16)sAraw[r][Ke];
      pk.h[1] = (__bf16)sAraw[r][Ke + 1];
      sAf[r >> 4][(r & 15) | (hf << 4)][v] = pk.u;
    }
    #pragma unroll
    for (int p = tid; p < 1024; p += 128) {         /* B: 16 Kpairs x 64 cols */
      int c  = p & 63;
      int Ke = (p >> 6) << 1;
      union { __bf16 h[2]; u32 u; } pk;
      pk.h[0] = (__bf16)sBraw[Ke][c];
      pk.h[1] = (__bf16)sBraw[Ke + 1][c];
      sBf[c >> 4][(c & 15) | (((Ke >> 3) & 1) << 4)][((Ke & 7) >> 1) | ((Ke >> 4) << 2)] = pk.u;
    }
    __syncthreads();

    v8u araw = *(const v8u*)&sAf[wave][lane][0];    /* 2x ds_load_b128 */
    v16bf af = __builtin_bit_cast(v16bf, araw);
    #pragma unroll
    for (int cb = 0; cb < 4; ++cb) {
      v8u braw = *(const v8u*)&sBf[cb][lane][0];
      v16bf bf = __builtin_bit_cast(v16bf, braw);
      acc[cb] = __builtin_amdgcn_wmma_f32_16x16x32_bf16(
          false, af, false, bf, (short)0, acc[cb], false, false);
    }
    __syncthreads();
  }

  /* C layout: vgpr r -> M = r + half*8, lane%16 -> N */
  #pragma unroll
  for (int cb = 0; cb < 4; ++cb) {
    const int gn = n0 + (cb << 4) + l16;
    #pragma unroll
    for (int r = 0; r < 8; ++r) {
      const int gm = m0 + (wave << 4) + r + (half << 3);
      float v = acc[cb][r];
      if (EP != EP_NONE)          v += bias[gn];
      if (EP == EP_BIAS_RELU)     v = fmaxf(v, 0.0f);
      if (EP == EP_BIAS_BN_RELU) { v = v*(bng[gn]*BN_SCALE) + bnb[gn]; v = fmaxf(v, 0.0f); }
      if (EP == EP_BIAS_RELU_BN) { v = fmaxf(v, 0.0f); v = v*(bng[gn]*BN_SCALE) + bnb[gn]; }
      if (EP == EP_BIAS_SIGMOID)  v = 1.0f/(1.0f + expf(-v));
      C[(long)gm*ldc + gn] = v;
    }
  }
}

/* -------------------------- elementwise --------------------------- */
__global__ __launch_bounds__(256) void k_fill(float* p, long n, float v) {
  long i = (long)blockIdx.x*blockDim.x + threadIdx.x;
  if (i < n) p[i] = v;
}

__global__ __launch_bounds__(256) void k_bn_rows(const float* __restrict__ x,
    const float* __restrict__ g, const float* __restrict__ b,
    float* __restrict__ y, long M, long N) {
  long i = (long)blockIdx.x*blockDim.x + threadIdx.x;
  if (i >= M*N) return;
  long c = i % N;
  y[i] = x[i]*(g[c]*BN_SCALE) + b[c];
}

/* -------------------------- graph path ---------------------------- */
__global__ __launch_bounds__(256) void k_edge_add(const float* __restrict__ X,
    const int* __restrict__ src, const int* __restrict__ dst,
    float* __restrict__ H, long E, int F) {
  long i = (long)blockIdx.x*blockDim.x + threadIdx.x;
  if (i >= E*(long)F) return;
  long e = i / F; int f = (int)(i - e*F);
  atomicAdd(&H[(long)dst[e]*F + f], X[(long)src[e]*F + f]);
}

__global__ __launch_bounds__(256) void k_deg(const int* __restrict__ dst,
                                             float* __restrict__ deg, long E) {
  long e = (long)blockIdx.x*blockDim.x + threadIdx.x;
  if (e < E) atomicAdd(&deg[dst[e]], 1.0f);
}

__global__ __launch_bounds__(256) void k_xw(const float* __restrict__ H,
    const float* __restrict__ w, float* __restrict__ xw, long n) {
  long i = (long)blockIdx.x*blockDim.x + threadIdx.x;
  if (i >= n) return;
  float s = 0.0f;
  for (int f = 0; f < 128; ++f) s += H[i*128 + f]*w[f];
  xw[i] = s;
}

__global__ __launch_bounds__(256) void k_edge_pool(const float* __restrict__ xw,
    const float* __restrict__ deg, const int* __restrict__ src,
    const int* __restrict__ dst, float* __restrict__ sacc, long E) {
  long e = (long)blockIdx.x*blockDim.x + threadIdx.x;
  if (e >= E) return;
  int s = src[e], d = dst[e];
  atomicAdd(&sacc[d], xw[s]*rsqrtf(deg[s]*deg[d]));
}

/* one workgroup per graph (64 contiguous nodes): softmax + weighted pool */
__global__ __launch_bounds__(128) void k_sagpool(const float* __restrict__ H,
    const float* __restrict__ xw, const float* __restrict__ sacc,
    const float* __restrict__ deg, const float* __restrict__ pbias,
    float* __restrict__ DSET, int slot) {
  int g = blockIdx.x, tid = threadIdx.x;
  __shared__ float sv[64];
  __shared__ float stat[2];
  long base = (long)g*64;
  if (tid < 64) {
    long n = base + tid;
    sv[tid] = tanhf(sacc[n] + xw[n]/deg[n] + pbias[0]);
  }
  __syncthreads();
  if (tid == 0) {
    float m = -1e30f;
    for (int n = 0; n < 64; ++n) m = fmaxf(m, sv[n]);
    float z = 0.0f;
    for (int n = 0; n < 64; ++n) z += expf(sv[n] - m);
    stat[0] = m; stat[1] = z;
  }
  __syncthreads();
  if (tid < 64) sv[tid] = expf(sv[tid] - stat[0])/stat[1];
  __syncthreads();
  float acc = 0.0f;
  for (int n = 0; n < 64; ++n) acc += H[(base + n)*128 + tid]*sv[n];
  DSET[(long)g*256 + slot*128 + tid] = acc;
}

/* -------------------------- mamba (per drug x bn-variant) --------- */
__global__ __launch_bounds__(256) void k_mamba(const int* __restrict__ smiles,
    const float* __restrict__ emb,
    const float* __restrict__ b1g, const float* __restrict__ b1b,
    const float* __restrict__ b2g, const float* __restrict__ b2b,
    const float* __restrict__ in_w, const float* __restrict__ conv_w,
    const float* __restrict__ conv_b, const float* __restrict__ xproj_w,
    const float* __restrict__ dt_w, const float* __restrict__ dt_b,
    const float* __restrict__ A_log, const float* __restrict__ Dp,
    const float* __restrict__ out_w,
    float* __restrict__ M1, float* __restrict__ M2) {
  int t = blockIdx.x*blockDim.x + threadIdx.x;
  if (t >= 1024) return;
  int g = t & 511, var = t >> 9;
  const float* bg = var ? b2g : b1g;
  const float* bb = var ? b2b : b1b;
  float* Mout = (var ? M2 : M1) + (long)g*256;

  float A[4][2], h[4][2] = {}, win[4][4] = {};
  #pragma unroll
  for (int d = 0; d < 4; ++d)
    for (int s = 0; s < 2; ++s) A[d][s] = -expf(A_log[d*2 + s]);

  for (int st = 0; st < 128; ++st) {
    int tok = smiles[(long)g*128 + st];
    float sc = bg[st]*BN_SCALE, of = bb[st];
    float x0 = emb[tok*2 + 0]*sc + of;
    float x1 = emb[tok*2 + 1]*sc + of;
    float xz[8];
    #pragma unroll
    for (int j = 0; j < 8; ++j) xz[j] = x0*in_w[j] + x1*in_w[8 + j];
    #pragma unroll
    for (int j = 0; j < 3; ++j)
      for (int d = 0; d < 4; ++d) win[j][d] = win[j+1][d];
    #pragma unroll
    for (int d = 0; d < 4; ++d) win[3][d] = xz[d];
    float xc[4];
    #pragma unroll
    for (int d = 0; d < 4; ++d) {
      float cv = conv_b[d];
      #pragma unroll
      for (int j = 0; j < 4; ++j) cv += win[j][d]*conv_w[d*4 + j];
      xc[d] = cv/(1.0f + expf(-cv));               /* silu */
    }
    float xdb[5];
    #pragma unroll
    for (int j = 0; j < 5; ++j) {
      float s = 0.0f;
      #pragma unroll
      for (int d = 0; d < 4; ++d) s += xc[d]*xproj_w[d*5 + j];
      xdb[j] = s;
    }
    float B0 = xdb[1], B1 = xdb[2], C0 = xdb[3], C1 = xdb[4];
    float o0 = 0.0f, o1 = 0.0f;
    #pragma unroll
    for (int d = 0; d < 4; ++d) {
      float dt = logf(1.0f + expf(xdb[0]*dt_w[d] + dt_b[d]));   /* softplus */
      h[d][0] = expf(dt*A[d][0])*h[d][0] + dt*B0*xc[d];
      h[d][1] = expf(dt*A[d][1])*h[d][1] + dt*B1*xc[d];
      float y = h[d][0]*C0 + h[d][1]*C1 + xc[d]*Dp[d];
      float z = xz[4 + d];
      y *= z/(1.0f + expf(-z));                    /* * silu(z) */
      o0 += y*out_w[d*2 + 0];
      o1 += y*out_w[d*2 + 1];
    }
    Mout[st*2 + 0] = o0;
    Mout[st*2 + 1] = o1;
  }
}

/* -------------------------- gathers / packs ----------------------- */
__global__ __launch_bounds__(256) void k_pack_sm(const float* __restrict__ M1,
    const float* __restrict__ M2, const int* __restrict__ dra,
    const int* __restrict__ drb, const float* __restrict__ g0,
    const float* __restrict__ b0, float* __restrict__ SMIN) {
  long i = (long)blockIdx.x*blockDim.x + threadIdx.x;
  if (i >= (long)BSZ*512) return;
  long b = i >> 9; int j = (int)(i & 511);
  float v = (j < 256) ? M1[(long)dra[b]*256 + j] : M2[(long)drb[b]*256 + (j - 256)];
  SMIN[i] = v*(g0[j]*BN_SCALE) + b0[j];
}

__global__ __launch_bounds__(256) void k_gather_cell(const float* __restrict__ C2,
    const int* __restrict__ cid, const float* __restrict__ g,
    const float* __restrict__ bb, float* __restrict__ SMCIN,
    float* __restrict__ GGCIN) {
  long i = (long)blockIdx.x*blockDim.x + threadIdx.x;
  if (i >= (long)BSZ*256) return;
  long b = i >> 8; int f = (int)(i & 255);
  float v = C2[(long)(cid[b] - 35)*256 + f]*(g[f]*BN_SCALE) + bb[f];
  SMCIN[b*768 + 512 + f] = v;
  GGCIN[b*768 + 512 + f] = v;
}

__global__ __launch_bounds__(256) void k_gather_dset(const float* __restrict__ DSET,
    const int* __restrict__ dra, const int* __restrict__ drb,
    float* __restrict__ D1, float* __restrict__ D2) {
  long i = (long)blockIdx.x*blockDim.x + threadIdx.x;
  if (i >= (long)BSZ*256) return;
  long b = i >> 8; int r = (int)(i & 255);
  D1[i] = DSET[(long)dra[b]*256 + r];
  D2[i] = DSET[(long)drb[b]*256 + r];
}

/* 2x2 per-head attention combine (L=2, dh=32, NH=4) */
__global__ __launch_bounds__(256) void k_attn(const float* __restrict__ Q,
    const float* __restrict__ K, const float* __restrict__ V,
    float* __restrict__ O) {
  int t = blockIdx.x*blockDim.x + threadIdx.x;
  if (t >= BSZ*4) return;
  int b = t >> 2, h = t & 3;
  const float* q0 = Q + ((long)b*2 + 0)*128 + h*32;
  const float* q1 = Q + ((long)b*2 + 1)*128 + h*32;
  const float* k0 = K + ((long)b*2 + 0)*128 + h*32;
  const float* k1 = K + ((long)b*2 + 1)*128 + h*32;
  const float* v0 = V + ((long)b*2 + 0)*128 + h*32;
  const float* v1 = V + ((long)b*2 + 1)*128 + h*32;
  float s00 = 0, s01 = 0, s10 = 0, s11 = 0;
  for (int d = 0; d < 32; ++d) {
    s00 += q0[d]*k0[d]; s01 += q0[d]*k1[d];
    s10 += q1[d]*k0[d]; s11 += q1[d]*k1[d];
  }
  const float inv = 0.17677669529663687f;   /* 1/sqrt(32) */
  s00 *= inv; s01 *= inv; s10 *= inv; s11 *= inv;
  float m0 = fmaxf(s00, s01), m1 = fmaxf(s10, s11);
  float e00 = expf(s00 - m0), e01 = expf(s01 - m0);
  float e10 = expf(s10 - m1), e11 = expf(s11 - m1);
  float a00 = e00/(e00 + e01), a01 = e01/(e00 + e01);
  float a10 = e10/(e10 + e11), a11 = e11/(e10 + e11);
  float* o0 = O + ((long)b*2 + 0)*128 + h*32;
  float* o1 = O + ((long)b*2 + 1)*128 + h*32;
  for (int d = 0; d < 32; ++d) {
    o0[d] = a00*v0[d] + a01*v1[d];
    o1[d] = a10*v0[d] + a11*v1[d];
  }
}

/* O = bn_axis1(O + R) */
__global__ __launch_bounds__(256) void k_residual_bn(float* __restrict__ O,
    const float* __restrict__ R, const float* __restrict__ g,
    const float* __restrict__ bb, long n) {
  long i = (long)blockIdx.x*blockDim.x + threadIdx.x;
  if (i >= n) return;
  int pos = (int)((i >> 7) & 1);
  O[i] = (O[i] + R[i])*(g[pos]*BN_SCALE) + bb[pos];
}

__global__ __launch_bounds__(256) void k_pack_gg(const float* __restrict__ O1L,
    const float* __restrict__ O2L, const float* __restrict__ g0,
    const float* __restrict__ b0, float* __restrict__ GG) {
  long i = (long)blockIdx.x*blockDim.x + threadIdx.x;
  if (i >= (long)BSZ*512) return;
  long b = i >> 9; int j = (int)(i & 511);
  int sel = j >> 7, f = j & 127, pos = sel & 1;
  const float* s = (sel < 2) ? O1L : O2L;
  float v = s[(b*2 + pos)*128 + f];
  GG[i] = v*(g0[j]*BN_SCALE) + b0[j];
}

__global__ __launch_bounds__(256) void k_highway(float* __restrict__ CAT,
    const float* __restrict__ GATE, const float* __restrict__ TR, long n) {
  long i = (long)blockIdx.x*blockDim.x + threadIdx.x;
  if (i >= n) return;
  float gt = GATE[i];
  CAT[i] = gt*TR[i] + (1.0f - gt)*CAT[i];
}

__global__ __launch_bounds__(256) void k_out(const float* __restrict__ H,
    const float* __restrict__ w, const float* __restrict__ bb,
    float* __restrict__ out) {
  int b = blockIdx.x*blockDim.x + threadIdx.x;
  if (b >= BSZ) return;
  float s = bb[0];
  for (int f = 0; f < 256; ++f) s += H[(long)b*256 + f]*w[f];
  out[b] = s;
}

/* ------------------------- input indexing ------------------------- */
enum {
  IN_DRUGF = 0, IN_ESRC, IN_EDST, IN_IBATCH, IN_GEXPR, IN_SMILES, IN_DRA, IN_DRB, IN_CID,
  P_BN_DRUG_G, P_BN_DRUG_B,
  P_GIN1_W1, P_GIN1_B1, P_GIN1_W2, P_GIN1_B2,
  P_BN_CONV1_G, P_BN_CONV1_B,
  P_GIN2_W1, P_GIN2_B1, P_GIN2_W2, P_GIN2_B2,
  P_BN_CONV2_G, P_BN_CONV2_B,
  P_POOL1_W, P_POOL1_B, P_POOL2_W, P_POOL2_B,
  P_BN_GEX_G, P_BN_GEX_B,
  P_CELL1_W, P_CELL1_B, P_BN_CELL1_G, P_BN_CELL1_B,
  P_CELL2_W, P_CELL2_B, P_BN_CELL2_G, P_BN_CELL2_B,
  P_BN_CELL_G, P_BN_CELL_B,
  P_EMB,
  P_BN_S1_G, P_BN_S1_B, P_BN_S2_G, P_BN_S2_B,
  P_M_INW, P_M_CONVW, P_M_CONVB, P_M_XPROJ, P_M_DTW, P_M_DTB, P_M_ALOG, P_M_D, P_M_OUTW,
  P_BN_SM0_G, P_BN_SM0_B,
  P_FCSM1_W, P_FCSM1_B, P_BN_SM1_G, P_BN_SM1_B,
  P_FCSMC_W, P_FCSMC_B, P_BN_SMC_G, P_BN_SMC_B,
  P_A1_WQ, P_A1_BQ, P_A1_WK, P_A1_BK, P_A1_WV, P_A1_BV, P_A1_WO, P_A1_BO,
  P_A2_WQ, P_A2_BQ, P_A2_WK, P_A2_BK, P_A2_WV, P_A2_BV, P_A2_WO, P_A2_BO,
  P_CABN1_G, P_CABN1_B, P_CABN2_G, P_CABN2_B,
  P_CALIN1_W, P_CALIN1_B, P_CALIN2_W, P_CALIN2_B,
  P_BN_G0_G, P_BN_G0_B,
  P_FCG1_W, P_FCG1_B, P_BN_G1_G, P_BN_G1_B,
  P_FCGC_W, P_FCGC_B, P_BN_GC_G, P_BN_GC_B,
  P_BN_CAT_G, P_BN_CAT_B,
  P_HWG_W, P_HWG_B, P_HWL_W, P_HWL_B, P_HWBN_G, P_HWBN_B,
  P_FC1_W, P_FC1_B, P_BN1_G, P_BN1_B,
  P_FCOUT_W, P_FCOUT_B
};

#define FPI(i) ((const float*)d_in[(i)])
#define IPI(i) ((const int*)d_in[(i)])
#define L1D(n) dim3((unsigned)(((n) + 255)/256)), dim3(256)

extern "C" void kernel_launch(void* const* d_in, const int* in_sizes, int n_in,
                              void* d_out, int out_size, void* d_ws, size_t ws_size,
                              hipStream_t stream) {
  (void)in_sizes; (void)n_in; (void)out_size; (void)ws_size;
  float* ws = (float*)d_ws;
  size_t off = 0;
  auto alloc = [&](size_t n) { float* p = ws + off; off += n; return p; };

  float* X0    = alloc((size_t)NNODE*78);
  float* HS    = alloc((size_t)NNODE*78);
  float* T     = alloc((size_t)NNODE*256);   /* reused for gin1/gin2 hidden */
  float* H1    = alloc((size_t)NNODE*128);
  float* HS2   = alloc((size_t)NNODE*128);
  float* H2    = alloc((size_t)NNODE*128);
  float* deg   = alloc(NNODE);
  float* xw    = alloc(NNODE);
  float* sacc  = alloc(NNODE);
  float* DSET  = alloc((size_t)GNUM*256);
  float* CBN   = alloc((size_t)128*954);
  float* C1    = alloc((size_t)128*512);
  float* C2    = alloc((size_t)128*256);
  float* M1    = alloc((size_t)GNUM*256);
  float* M2    = alloc((size_t)GNUM*256);
  float* SMIN  = alloc((size_t)BSZ*512);
  float* SMCIN = alloc((size_t)BSZ*768);
  float* D1    = alloc((size_t)BSZ*256);
  float* D2    = alloc((size_t)BSZ*256);
  float* Qb    = alloc((size_t)BSZ*256);
  float* Kb    = alloc((size_t)BSZ*256);
  float* Vb    = alloc((size_t)BSZ*256);
  float* AO    = alloc((size_t)BSZ*256);
  float* O1    = alloc((size_t)BSZ*256);
  float* O1L   = alloc((size_t)BSZ*256);
  float* O2    = alloc((size_t)BSZ*256);
  float* O2L   = alloc((size_t)BSZ*256);
  float* GG    = alloc((size_t)BSZ*512);
  float* GGCIN = alloc((size_t)BSZ*768);
  float* CAT   = alloc((size_t)BSZ*1536);
  float* GATE  = alloc((size_t)BSZ*1536);
  float* TR    = alloc((size_t)BSZ*1536);
  float* HF    = alloc((size_t)BSZ*256);

  auto gg = [](int M, int N) { return dim3((unsigned)(N/64), (unsigned)(M/64)); };

  /* ---- drug graph path ---- */
  k_bn_rows<<<L1D((long)NNODE*78), 0, stream>>>(FPI(IN_DRUGF), FPI(P_BN_DRUG_G), FPI(P_BN_DRUG_B), X0, NNODE, 78);
  hipMemcpyAsync(HS, X0, sizeof(float)*(size_t)NNODE*78, hipMemcpyDeviceToDevice, stream);
  k_edge_add<<<L1D((long)NEDGE*78), 0, stream>>>(X0, IPI(IN_ESRC), IPI(IN_EDST), HS, NEDGE, 78);
  k_gemm<EP_BIAS_RELU><<<gg(NNODE,256), 128, 0, stream>>>(HS, 78, FPI(P_GIN1_W1), T, 256, NNODE, 256, 78, FPI(P_GIN1_B1), nullptr, nullptr);
  k_gemm<EP_BIAS_RELU_BN><<<gg(NNODE,128), 128, 0, stream>>>(T, 256, FPI(P_GIN1_W2), H1, 128, NNODE, 128, 256, FPI(P_GIN1_B2), FPI(P_BN_CONV1_G), FPI(P_BN_CONV1_B));
  hipMemcpyAsync(HS2, H1, sizeof(float)*(size_t)NNODE*128, hipMemcpyDeviceToDevice, stream);
  k_edge_add<<<L1D((long)NEDGE*128), 0, stream>>>(H1, IPI(IN_ESRC), IPI(IN_EDST), HS2, NEDGE, 128);
  k_gemm<EP_BIAS_RELU><<<gg(NNODE,256), 128, 0, stream>>>(HS2, 128, FPI(P_GIN2_W1), T, 256, NNODE, 256, 128, FPI(P_GIN2_B1), nullptr, nullptr);
  k_gemm<EP_BIAS_RELU_BN><<<gg(NNODE,128), 128, 0, stream>>>(T, 256, FPI(P_GIN2_W2), H2, 128, NNODE, 128, 256, FPI(P_GIN2_B2), FPI(P_BN_CONV2_G), FPI(P_BN_CONV2_B));

  /* degrees + SAGPool 1/2 */
  k_fill<<<L1D(NNODE), 0, stream>>>(deg, NNODE, 1.0f);
  k_deg<<<L1D(NEDGE), 0, stream>>>(IPI(IN_EDST), deg, NEDGE);
  k_xw<<<L1D(NNODE), 0, stream>>>(H1, FPI(P_POOL1_W), xw, NNODE);
  k_fill<<<L1D(NNODE), 0, stream>>>(sacc, NNODE, 0.0f);
  k_edge_pool<<<L1D(NEDGE), 0, stream>>>(xw, deg, IPI(IN_ESRC), IPI(IN_EDST), sacc, NEDGE);
  k_sagpool<<<GNUM, 128, 0, stream>>>(H1, xw, sacc, deg, FPI(P_POOL1_B), DSET, 0);
  k_xw<<<L1D(NNODE), 0, stream>>>(H2, FPI(P_POOL2_W), xw, NNODE);
  k_fill<<<L1D(NNODE), 0, stream>>>(sacc, NNODE, 0.0f);
  k_edge_pool<<<L1D(NEDGE), 0, stream>>>(xw, deg, IPI(IN_ESRC), IPI(IN_EDST), sacc, NEDGE);
  k_sagpool<<<GNUM, 128, 0, stream>>>(H2, xw, sacc, deg, FPI(P_POOL2_B), DSET, 1);

  /* ---- cell line path ---- */
  k_bn_rows<<<L1D((long)128*954), 0, stream>>>(FPI(IN_GEXPR), FPI(P_BN_GEX_G), FPI(P_BN_GEX_B), CBN, 128, 954);
  k_gemm<EP_BIAS_BN_RELU><<<gg(128,512), 128, 0, stream>>>(CBN, 954, FPI(P_CELL1_W), C1, 512, 128, 512, 954, FPI(P_CELL1_B), FPI(P_BN_CELL1_G), FPI(P_BN_CELL1_B));
  k_gemm<EP_BIAS_BN_RELU><<<gg(128,256), 128, 0, stream>>>(C1, 512, FPI(P_CELL2_W), C2, 256, 128, 256, 512, FPI(P_CELL2_B), FPI(P_BN_CELL2_G), FPI(P_BN_CELL2_B));

  /* ---- mamba (dedup: 512 drugs x 2 bn variants) ---- */
  k_mamba<<<4, 256, 0, stream>>>(IPI(IN_SMILES), FPI(P_EMB),
      FPI(P_BN_S1_G), FPI(P_BN_S1_B), FPI(P_BN_S2_G), FPI(P_BN_S2_B),
      FPI(P_M_INW), FPI(P_M_CONVW), FPI(P_M_CONVB), FPI(P_M_XPROJ),
      FPI(P_M_DTW), FPI(P_M_DTB), FPI(P_M_ALOG), FPI(P_M_D), FPI(P_M_OUTW), M1, M2);

  /* ---- smiles branch ---- */
  k_pack_sm<<<L1D((long)BSZ*512), 0, stream>>>(M1, M2, IPI(IN_DRA), IPI(IN_DRB), FPI(P_BN_SM0_G), FPI(P_BN_SM0_B), SMIN);
  k_gemm<EP_BIAS_BN_RELU><<<gg(BSZ,512), 128, 0, stream>>>(SMIN, 512, FPI(P_FCSM1_W), SMCIN, 768, BSZ, 512, 512, FPI(P_FCSM1_B), FPI(P_BN_SM1_G), FPI(P_BN_SM1_B));
  k_gather_cell<<<L1D((long)BSZ*256), 0, stream>>>(C2, IPI(IN_CID), FPI(P_BN_CELL_G), FPI(P_BN_CELL_B), SMCIN, GGCIN);
  k_gemm<EP_BIAS_BN_RELU><<<gg(BSZ,768), 128, 0, stream>>>(SMCIN, 768, FPI(P_FCSMC_W), CAT, 1536, BSZ, 768, 768, FPI(P_FCSMC_B), FPI(P_BN_SMC_G), FPI(P_BN_SMC_B));

  /* ---- cross attention branch ---- */
  k_gather_dset<<<L1D((long)BSZ*256), 0, stream>>>(DSET, IPI(IN_DRA), IPI(IN_DRB), D1, D2);
  /* attn1: q=d1, k=v=d2 */
  k_gemm<EP_BIAS><<<gg(2*BSZ,128), 128, 0, stream>>>(D1, 128, FPI(P_A1_WQ), Qb, 128, 2*BSZ, 128, 128, FPI(P_A1_BQ), nullptr, nullptr);
  k_gemm<EP_BIAS><<<gg(2*BSZ,128), 128, 0, stream>>>(D2, 128, FPI(P_A1_WK), Kb, 128, 2*BSZ, 128, 128, FPI(P_A1_BK), nullptr, nullptr);
  k_gemm<EP_BIAS><<<gg(2*BSZ,128), 128, 0, stream>>>(D2, 128, FPI(P_A1_WV), Vb, 128, 2*BSZ, 128, 128, FPI(P_A1_BV), nullptr, nullptr);
  k_attn<<<L1D(BSZ*4), 0, stream>>>(Qb, Kb, Vb, AO);
  k_gemm<EP_BIAS><<<gg(2*BSZ,128), 128, 0, stream>>>(AO, 128, FPI(P_A1_WO), O1, 128, 2*BSZ, 128, 128, FPI(P_A1_BO), nullptr, nullptr);
  k_residual_bn<<<L1D((long)2*BSZ*128), 0, stream>>>(O1, D1, FPI(P_CABN1_G), FPI(P_CABN1_B), (long)2*BSZ*128);
  k_gemm<EP_BIAS_RELU><<<gg(2*BSZ,128), 128, 0, stream>>>(O1, 128, FPI(P_CALIN1_W), O1L, 128, 2*BSZ, 128, 128, FPI(P_CALIN1_B), nullptr, nullptr);
  /* attn2: q=d2, k=v=d1 */
  k_gemm<EP_BIAS><<<gg(2*BSZ,128), 128, 0, stream>>>(D2, 128, FPI(P_A2_WQ), Qb, 128, 2*BSZ, 128, 128, FPI(P_A2_BQ), nullptr, nullptr);
  k_gemm<EP_BIAS><<<gg(2*BSZ,128), 128, 0, stream>>>(D1, 128, FPI(P_A2_WK), Kb, 128, 2*BSZ, 128, 128, FPI(P_A2_BK), nullptr, nullptr);
  k_gemm<EP_BIAS><<<gg(2*BSZ,128), 128, 0, stream>>>(D1, 128, FPI(P_A2_WV), Vb, 128, 2*BSZ, 128, 128, FPI(P_A2_BV), nullptr, nullptr);
  k_attn<<<L1D(BSZ*4), 0, stream>>>(Qb, Kb, Vb, AO);
  k_gemm<EP_BIAS><<<gg(2*BSZ,128), 128, 0, stream>>>(AO, 128, FPI(P_A2_WO), O2, 128, 2*BSZ, 128, 128, FPI(P_A2_BO), nullptr, nullptr);
  k_residual_bn<<<L1D((long)2*BSZ*128), 0, stream>>>(O2, D2, FPI(P_CABN2_G), FPI(P_CABN2_B), (long)2*BSZ*128);
  k_gemm<EP_BIAS_RELU><<<gg(2*BSZ,128), 128, 0, stream>>>(O2, 128, FPI(P_CALIN2_W), O2L, 128, 2*BSZ, 128, 128, FPI(P_CALIN2_B), nullptr, nullptr);

  k_pack_gg<<<L1D((long)BSZ*512), 0, stream>>>(O1L, O2L, FPI(P_BN_G0_G), FPI(P_BN_G0_B), GG);
  k_gemm<EP_BIAS_BN_RELU><<<gg(BSZ,512), 128, 0, stream>>>(GG, 512, FPI(P_FCG1_W), GGCIN, 768, BSZ, 512, 512, FPI(P_FCG1_B), FPI(P_BN_G1_G), FPI(P_BN_G1_B));
  k_gemm<EP_BIAS_BN_RELU><<<gg(BSZ,768), 128, 0, stream>>>(GGCIN, 768, FPI(P_FCGC_W), CAT + 768, 1536, BSZ, 768, 768, FPI(P_FCGC_B), FPI(P_BN_GC_G), FPI(P_BN_GC_B));

  /* ---- head: bn_cat, highway, fc1, fc_out ---- */
  k_bn_rows<<<L1D((long)BSZ*1536), 0, stream>>>(CAT, FPI(P_BN_CAT_G), FPI(P_BN_CAT_B), CAT, BSZ, 1536);
  k_gemm<EP_BIAS_SIGMOID><<<gg(BSZ,1536), 128, 0, stream>>>(CAT, 1536, FPI(P_HWG_W), GATE, 1536, BSZ, 1536, 1536, FPI(P_HWG_B), nullptr, nullptr);
  k_gemm<EP_BIAS_RELU_BN><<<gg(BSZ,1536), 128, 0, stream>>>(CAT, 1536, FPI(P_HWL_W), TR, 1536, BSZ, 1536, 1536, FPI(P_HWL_B), FPI(P_HWBN_G), FPI(P_HWBN_B));
  k_highway<<<L1D((long)BSZ*1536), 0, stream>>>(CAT, GATE, TR, (long)BSZ*1536);
  k_gemm<EP_BIAS_BN_RELU><<<gg(BSZ,256), 128, 0, stream>>>(CAT, 1536, FPI(P_FC1_W), HF, 256, BSZ, 256, 1536, FPI(P_FC1_B), FPI(P_BN1_G), FPI(P_BN1_B));
  k_out<<<L1D(BSZ), 0, stream>>>(HF, FPI(P_FCOUT_W), FPI(P_FCOUT_B), (float*)d_out);
}
